// Autoencoder_61125974556730
// MI455X (gfx1250) — matrix operand order repaired and checked by reading, not compiled, not run on hardware
//
#include <hip/hip_runtime.h>

#define DEV __device__ __forceinline__

typedef __attribute__((ext_vector_type(16))) __bf16 v16bf;
typedef __attribute__((ext_vector_type(8)))  __bf16 v8bf;
typedef __attribute__((ext_vector_type(8)))  float  v8f;

constexpr int B = 32, T = 64, S = 128, D = 512, V = 32000;

// ---------------- WMMA tile helpers (layouts per CDNA5 ISA 7.12.2) ----------------

// A: 16-bit matrix 16x32 (MxK), row-major source with leading dim lda.
// lanes 0-15 : M=lane,    elems 0-7 = K k0..k0+7,   elems 8-15 = K k0+16..k0+23
// lanes 16-31: M=lane-16, elems 0-7 = K k0+8..k0+15, elems 8-15 = K k0+24..k0+31
DEV v16bf load_a16(const __bf16* A, int lda, int k0, int lane) {
  int half = (lane >> 4) & 1;
  int m    = lane & 15;
  const __bf16* p = A + (size_t)m * lda + k0 + half * 8;
  v8bf lo = *(const v8bf*)(p);
  v8bf hi = *(const v8bf*)(p + 16);
  return __builtin_shufflevector(lo, hi, 0,1,2,3,4,5,6,7,8,9,10,11,12,13,14,15);
}

// B: 16-bit matrix 32x16 (KxN). Source is N-major (weight row n contiguous in k):
// Bm[n*ldb + k]. lanes 0-15: N=lane, K k0..k0+7, k0+16..k0+23 ; lanes 16-31: N=lane-16, K k0+8..k0+15, k0+24..k0+31.
DEV v16bf load_b16(const __bf16* Bm, int ldb, int k0, int lane) {
  int half = (lane >> 4) & 1;
  int n    = lane & 15;
  const __bf16* p = Bm + (size_t)n * ldb + k0 + half * 8;
  v8bf lo = *(const v8bf*)(p);
  v8bf hi = *(const v8bf*)(p + 16);
  return __builtin_shufflevector(lo, hi, 0,1,2,3,4,5,6,7,8,9,10,11,12,13,14,15);
}

DEV v8f wmma_bf16(v16bf a, v16bf b, v8f c) {
  // (neg_a, A, neg_b, B, c_mod, C, reuse_a, reuse_b)
  return __builtin_amdgcn_wmma_f32_16x16x32_bf16(false, a, false, b, (short)0, c,
                                                 false, false);
}

DEV float sigmoidf(float x) { return 1.f / (1.f + __expf(-x)); }

// ---------------- pre-pass kernels ----------------

__global__ void k_cvt_bf16(const float* __restrict__ src, __bf16* __restrict__ dst, int n) {
  int i = blockIdx.x * 256 + threadIdx.x;
  if (i < n) dst[i] = (__bf16)src[i];
}

__global__ void k_embed(const int* __restrict__ words, const float* __restrict__ emb,
                        __bf16* __restrict__ x_bf) {
  int i = blockIdx.x * 256 + threadIdx.x;           // over B*T*D
  if (i >= B * T * D) return;
  int d  = i % D;
  int bt = i / D;                                   // b*T + t
  x_bf[i] = (__bf16)emb[(size_t)words[bt] * D + d];
}

__global__ void k_init_state(const int* __restrict__ input_len,
                             const float* __restrict__ pre_h,
                             const float* __restrict__ cell0,
                             float* __restrict__ hid_f, __bf16* __restrict__ hid_bf,
                             float* __restrict__ cell_f, __bf16* __restrict__ cell_bf) {
  int i = blockIdx.x * 256 + threadIdx.x;           // over B*D
  if (i >= B * D) return;
  int b = i / D, d = i % D;
  float h = pre_h[(((size_t)input_len[b] - 1) * B + b) * D + d];
  hid_f[i]  = h;  hid_bf[i]  = (__bf16)h;
  float c   = cell0[i];
  cell_f[i] = c;  cell_bf[i] = (__bf16)c;
}

// enc_proj[s,b,e] = sum_d pre_h[s,b,d] * W_h[e,d]   (M = S*B = 4096, N = D, K = D)
__global__ void k_encproj(const __bf16* __restrict__ A, const __bf16* __restrict__ W,
                          float* __restrict__ C) {
  int lane = threadIdx.x;
  int n0 = blockIdx.x * 16;
  int m0 = blockIdx.y * 16;
  v8f acc = {};
  for (int k = 0; k < D; k += 32)
    acc = wmma_bf16(load_a16(A + (size_t)m0 * D, D, k, lane),
                    load_b16(W + (size_t)n0 * D, D, k, lane), acc);
  int half = lane >> 4, n = lane & 15;
#pragma unroll
  for (int r = 0; r < 8; ++r)
    C[(size_t)(m0 + half * 8 + r) * D + n0 + n] = acc[r];
}

// ---------------- per-step kernels ----------------

// Fused LSTM cell: gates = x_t@W_ih^T + b_ih + h@W_hh^T + b_hh, then pointwise update.
// Block j owns gate columns [16j,16j+16); computes all 4 gate groups so the
// pointwise part is local. grid = D/16, block = 32 (one wave).
__global__ void k_lstm(const __bf16* __restrict__ xt, // = x_bf + t*D, lda = T*D
                       const __bf16* __restrict__ hprev_bf,
                       const float*  __restrict__ hprev_f,
                       const __bf16* __restrict__ Wih, const __bf16* __restrict__ Whh,
                       const float* __restrict__ bih, const float* __restrict__ bhh,
                       const int* __restrict__ lengths, int t,
                       float* __restrict__ cell_f, __bf16* __restrict__ cell_bf,
                       float* __restrict__ hnew_f, __bf16* __restrict__ hnew_bf) {
  int lane = threadIdx.x;
  int jc = blockIdx.x * 16;
  v8f acc[2][4] = {};
  for (int k = 0; k < D; k += 32) {                 // x contribution
    v16bf a0 = load_a16(xt, T * D, k, lane);
    v16bf a1 = load_a16(xt + (size_t)16 * T * D, T * D, k, lane);
#pragma unroll
    for (int g = 0; g < 4; ++g) {
      v16bf bm = load_b16(Wih + (size_t)(g * D + jc) * D, D, k, lane);
      acc[0][g] = wmma_bf16(a0, bm, acc[0][g]);
      acc[1][g] = wmma_bf16(a1, bm, acc[1][g]);
    }
  }
  for (int k = 0; k < D; k += 32) {                 // h contribution
    v16bf a0 = load_a16(hprev_bf, D, k, lane);
    v16bf a1 = load_a16(hprev_bf + (size_t)16 * D, D, k, lane);
#pragma unroll
    for (int g = 0; g < 4; ++g) {
      v16bf bm = load_b16(Whh + (size_t)(g * D + jc) * D, D, k, lane);
      acc[0][g] = wmma_bf16(a0, bm, acc[0][g]);
      acc[1][g] = wmma_bf16(a1, bm, acc[1][g]);
    }
  }
  int half = lane >> 4, n = lane & 15;
  int c = jc + n;
#pragma unroll
  for (int mt = 0; mt < 2; ++mt) {
#pragma unroll
    for (int r = 0; r < 8; ++r) {
      int b = mt * 16 + half * 8 + r;
      float gi = acc[mt][0][r] + bih[0 * D + c] + bhh[0 * D + c];
      float gf = acc[mt][1][r] + bih[1 * D + c] + bhh[1 * D + c];
      float gg = acc[mt][2][r] + bih[2 * D + c] + bhh[2 * D + c];
      float go = acc[mt][3][r] + bih[3 * D + c] + bhh[3 * D + c];
      float i_ = sigmoidf(gi), f_ = sigmoidf(gf), o_ = sigmoidf(go);
      float g_ = tanhf(gg);
      float cold = cell_f[(size_t)b * D + c];
      float cnew = f_ * cold + i_ * g_;
      float hnew = o_ * tanhf(cnew);
      bool  msk  = t < lengths[b];
      float hold = hprev_f[(size_t)b * D + c];
      float ccur = msk ? cnew : cold;
      float hcur = msk ? hnew : hold;
      cell_f [(size_t)b * D + c] = ccur;
      cell_bf[(size_t)b * D + c] = (__bf16)ccur;
      hnew_f [(size_t)b * D + c] = hcur;
      hnew_bf[(size_t)b * D + c] = (__bf16)hcur;
    }
  }
}

// q = cell @ W_s^T + b_s   (M=32, N=D, K=D), fp32 out
__global__ void k_qproj(const __bf16* __restrict__ A, const __bf16* __restrict__ W,
                        const float* __restrict__ bias, float* __restrict__ C) {
  int lane = threadIdx.x;
  int n0 = blockIdx.x * 16;
  int m0 = blockIdx.y * 16;
  v8f acc = {};
  for (int k = 0; k < D; k += 32)
    acc = wmma_bf16(load_a16(A + (size_t)m0 * D, D, k, lane),
                    load_b16(W + (size_t)n0 * D, D, k, lane), acc);
  int half = lane >> 4, n = lane & 15;
#pragma unroll
  for (int r = 0; r < 8; ++r)
    C[(size_t)(m0 + half * 8 + r) * D + n0 + n] = acc[r] + bias[n0 + n];
}

// attention: scores -> softmax over S -> context ; one block per batch element
__global__ void k_attn(const float* __restrict__ enc_proj, const float* __restrict__ q,
                       const float* __restrict__ v_t, const float* __restrict__ pre_h,
                       const int* __restrict__ input_len, __bf16* __restrict__ ctx_bf) {
  int b = blockIdx.x;
  int tid = threadIdx.x;                 // 128 threads, thread = source position s
  int L = input_len[b];
  const float4* ep4 = (const float4*)(enc_proj + ((size_t)tid * B + b) * D);
  const float4* q4  = (const float4*)(q + (size_t)b * D);
  const float4* vt4 = (const float4*)v_t;
  float acc = 0.f;
  for (int d = 0; d < D / 4; ++d) {
    float4 e = ep4[d], qq = q4[d], vv = vt4[d];
    acc += vv.x * tanhf(e.x + qq.x) + vv.y * tanhf(e.y + qq.y) +
           vv.z * tanhf(e.z + qq.z) + vv.w * tanhf(e.w + qq.w);
  }
  float sc = (tid < L) ? acc : -INFINITY;
  __shared__ float red[S];
  __shared__ float aw[S];
  red[tid] = sc; __syncthreads();
  for (int off = 64; off > 0; off >>= 1) {
    if (tid < off) red[tid] = fmaxf(red[tid], red[tid + off]);
    __syncthreads();
  }
  float mx = red[0]; __syncthreads();
  float e = (tid < L) ? __expf(sc - mx) : 0.f;
  red[tid] = e; __syncthreads();
  for (int off = 64; off > 0; off >>= 1) {
    if (tid < off) red[tid] += red[tid + off];
    __syncthreads();
  }
  float inv = 1.f / red[0]; __syncthreads();
  aw[tid] = e * inv; __syncthreads();
  for (int d = tid; d < D; d += 128) {
    float c = 0.f;
    for (int ss = 0; ss < S; ++ss)
      c += aw[ss] * pre_h[((size_t)ss * B + b) * D + d];
    ctx_bf[(size_t)b * D + d] = (__bf16)c;
  }
}

// feat = [context, cell] @ V_w^T + V_b  (M=32, N=D, K=2D), bf16 out
__global__ void k_feat(const __bf16* __restrict__ ctx, const __bf16* __restrict__ cell,
                       const __bf16* __restrict__ Vw, const float* __restrict__ Vb,
                       __bf16* __restrict__ feat) {
  int lane = threadIdx.x;
  int n0 = blockIdx.x * 16;
  int m0 = blockIdx.y * 16;
  v8f acc = {};
  for (int k = 0; k < D; k += 32)
    acc = wmma_bf16(load_a16(ctx + (size_t)m0 * D, D, k, lane),
                    load_b16(Vw + (size_t)n0 * 2 * D, 2 * D, k, lane), acc);
  for (int k = 0; k < D; k += 32)
    acc = wmma_bf16(load_a16(cell + (size_t)m0 * D, D, k, lane),
                    load_b16(Vw + (size_t)n0 * 2 * D, 2 * D, k + D, lane), acc);
  int half = lane >> 4, n = lane & 15;
#pragma unroll
  for (int r = 0; r < 8; ++r)
    feat[(size_t)(m0 + half * 8 + r) * D + n0 + n] = (__bf16)(acc[r] + Vb[n0 + n]);
}

// out[t,b,:] = mask ? feat @ Vp_w^T + Vp_b : 0  (M=32, N=V=32000, K=D)
// Each block: 64 vocab columns (4 N-tiles) x full M; A tiles reused across N tiles.
__global__ void k_outproj(const __bf16* __restrict__ feat, const __bf16* __restrict__ Wp,
                          const float* __restrict__ bp, const int* __restrict__ lengths,
                          int t, float* __restrict__ out) {
  int lane = threadIdx.x;
  int N0 = blockIdx.x * 64;
  v8f acc[2][4] = {};
  for (int k = 0; k < D; k += 32) {
    v16bf a0 = load_a16(feat, D, k, lane);
    v16bf a1 = load_a16(feat + (size_t)16 * D, D, k, lane);
#pragma unroll
    for (int nt = 0; nt < 4; ++nt) {
      v16bf bm = load_b16(Wp + (size_t)(N0 + nt * 16) * D, D, k, lane);
      acc[0][nt] = wmma_bf16(a0, bm, acc[0][nt]);
      acc[1][nt] = wmma_bf16(a1, bm, acc[1][nt]);
    }
  }
  int half = lane >> 4, n = lane & 15;
#pragma unroll
  for (int mt = 0; mt < 2; ++mt) {
#pragma unroll
    for (int nt = 0; nt < 4; ++nt) {
#pragma unroll
      for (int r = 0; r < 8; ++r) {
        int b = mt * 16 + half * 8 + r;
        int v = N0 + nt * 16 + n;
        float val = (t < lengths[b]) ? (acc[mt][nt][r] + bp[v]) : 0.f;
        out[((size_t)t * B + b) * V + v] = val;
      }
    }
  }
}

// ---------------- host orchestration ----------------

extern "C" void kernel_launch(void* const* d_in, const int* in_sizes, int n_in,
                              void* d_out, int out_size, void* d_ws, size_t ws_size,
                              hipStream_t stream) {
  const int*   words     = (const int*)  d_in[0];
  const int*   lengths   = (const int*)  d_in[1];
  const int*   input_len = (const int*)  d_in[2];
  const float* pre_h     = (const float*)d_in[3];
  const float* cell0     = (const float*)d_in[4];
  const float* emb       = (const float*)d_in[5];
  const float* W_ih      = (const float*)d_in[6];
  const float* W_hh      = (const float*)d_in[7];
  const float* b_ih      = (const float*)d_in[8];
  const float* b_hh      = (const float*)d_in[9];
  const float* W_h       = (const float*)d_in[10];
  const float* W_s       = (const float*)d_in[11];
  const float* b_s       = (const float*)d_in[12];
  const float* v_t       = (const float*)d_in[13];
  const float* V_w       = (const float*)d_in[14];
  const float* V_b       = (const float*)d_in[15];
  const float* Vp_w      = (const float*)d_in[16];
  const float* Vp_b      = (const float*)d_in[17];
  float* out = (float*)d_out;

  // workspace carve-out (256B aligned)
  char* ws = (char*)d_ws;
  auto alloc = [&](size_t bytes) {
    char* p = ws;
    ws += (bytes + 255) & ~(size_t)255;
    return p;
  };
  __bf16* Vp_bf   = (__bf16*)alloc((size_t)V * D * 2);
  __bf16* Wih_bf  = (__bf16*)alloc((size_t)4 * D * D * 2);
  __bf16* Whh_bf  = (__bf16*)alloc((size_t)4 * D * D * 2);
  __bf16* Ws_bf   = (__bf16*)alloc((size_t)D * D * 2);
  __bf16* Vw_bf   = (__bf16*)alloc((size_t)D * 2 * D * 2);
  __bf16* Wh_bf   = (__bf16*)alloc((size_t)D * D * 2);
  __bf16* x_bf    = (__bf16*)alloc((size_t)B * T * D * 2);
  __bf16* preh_bf = (__bf16*)alloc((size_t)S * B * D * 2);
  float*  encp    = (float*) alloc((size_t)S * B * D * 4);
  float*  hid_f[2];  __bf16* hid_bf[2];
  hid_f[0]  = (float*) alloc((size_t)B * D * 4);
  hid_f[1]  = (float*) alloc((size_t)B * D * 4);
  hid_bf[0] = (__bf16*)alloc((size_t)B * D * 2);
  hid_bf[1] = (__bf16*)alloc((size_t)B * D * 2);
  float*  cell_f  = (float*) alloc((size_t)B * D * 4);
  __bf16* cell_bf = (__bf16*)alloc((size_t)B * D * 2);
  float*  q_f     = (float*) alloc((size_t)B * D * 4);
  __bf16* ctx_bf  = (__bf16*)alloc((size_t)B * D * 2);
  __bf16* feat_bf = (__bf16*)alloc((size_t)B * D * 2);

  auto cvt = [&](const float* s, __bf16* dst, size_t n) {
    k_cvt_bf16<<<(int)((n + 255) / 256), 256, 0, stream>>>(s, dst, (int)n);
  };

  // one-time conversions / precompute
  cvt(Vp_w, Vp_bf, (size_t)V * D);
  cvt(W_ih, Wih_bf, (size_t)4 * D * D);
  cvt(W_hh, Whh_bf, (size_t)4 * D * D);
  cvt(W_s,  Ws_bf,  (size_t)D * D);
  cvt(V_w,  Vw_bf,  (size_t)D * 2 * D);
  cvt(W_h,  Wh_bf,  (size_t)D * D);
  cvt(pre_h, preh_bf, (size_t)S * B * D);
  k_embed<<<(B * T * D + 255) / 256, 256, 0, stream>>>(words, emb, x_bf);
  k_init_state<<<(B * D + 255) / 256, 256, 0, stream>>>(input_len, pre_h, cell0,
                                                        hid_f[0], hid_bf[0],
                                                        cell_f, cell_bf);
  k_encproj<<<dim3(D / 16, (S * B) / 16), 32, 0, stream>>>(preh_bf, Wh_bf, encp);

  // sequential decode
  int cur = 0;
  for (int t = 0; t < T; ++t) {
    int nxt = cur ^ 1;
    k_lstm<<<D / 16, 32, 0, stream>>>(x_bf + (size_t)t * D, hid_bf[cur], hid_f[cur],
                                      Wih_bf, Whh_bf, b_ih, b_hh, lengths, t,
                                      cell_f, cell_bf, hid_f[nxt], hid_bf[nxt]);
    k_qproj<<<dim3(D / 16, B / 16), 32, 0, stream>>>(cell_bf, Ws_bf, b_s, q_f);
    k_attn<<<B, S, 0, stream>>>(encp, q_f, v_t, pre_h, input_len, ctx_bf);
    k_feat<<<dim3(D / 16, B / 16), 32, 0, stream>>>(ctx_bf, cell_bf, Vw_bf, V_b,
                                                    feat_bf);
    k_outproj<<<V / 64, 32, 0, stream>>>(feat_bf, Vp_bf, Vp_b, lengths, t, out);
    cur = nxt;
  }
  (void)in_sizes; (void)n_in; (void)out_size; (void)ws_size;
}
